// rn_layer_class_86414741996130
// MI455X (gfx1250) — matrix operand order repaired and checked by reading, not compiled, not run on hardware
//
#include <hip/hip_runtime.h>
#include <hip/hip_bf16.h>

typedef _Float16 v16h __attribute__((ext_vector_type(16)));
typedef _Float16 v8h  __attribute__((ext_vector_type(8)));
typedef float    v8f  __attribute__((ext_vector_type(8)));
typedef unsigned int u32x4 __attribute__((ext_vector_type(4)));
typedef int          i32x8 __attribute__((ext_vector_type(8)));
typedef int          i32x4 __attribute__((ext_vector_type(4)));

#define MB      16
#define NUM     128
#define LENTH   64
#define SIZE    256
#define OUTC    256
#define DIN     130        // 2*(LENTH+1)
#define K1PAD   160        // DIN padded to multiple of 32
#define KC1     5          // K1PAD/32
#define KC2     8          // SIZE/32
#define NT      16         // SIZE/16 N-tiles
#define ROWS_PER_WG 256
#define THREADS 256
#define TILES_PER_WAVE 2
#define HSTRIDE 264        // lds_h row stride in halfs (256 + 8 pad)
#define LDSH_BYTES (ROWS_PER_WG * HSTRIDE * 2)

// fragment-packed weight offsets in d_ws (in halfs)
#define W1F_OFF 0
#define W1F_CNT (NT*KC1*512)
#define W2F_OFF (W1F_OFF + W1F_CNT)
#define W2F_CNT (NT*KC2*512)
#define W3F_OFF (W2F_OFF + W2F_CNT)
#define W4F_OFF (W3F_OFF + W2F_CNT)
#define WF_TOTAL (W4F_OFF + W2F_CNT)   // 237568 halfs = 475136 bytes

// ---------------------------------------------------------------------------
// Repack f32 weights into f16 WMMA-B fragments.
// B layout (16x16x32 f16, wave32): fragment = 512 halfs, element (lane, h):
//   N = n0 + (lane & 15),  K = k0 + ((h>>3)<<4) + ((lane>>4)<<3) + (h&7)   (h = 0..15)
// Fragment order: [nt][kc], each 512 halfs, lane-major (lane*16 + h).
// ---------------------------------------------------------------------------
__global__ void rn_pack_weights(const float* __restrict__ W1,
                                const float* __restrict__ W2,
                                const float* __restrict__ W3,
                                const float* __restrict__ W4,
                                _Float16* __restrict__ wfrag) {
  int idx = blockIdx.x * blockDim.x + threadIdx.x;
  if (idx >= WF_TOTAL) return;
  const float* W; int kcn, rem, kdim;
  if (idx < W2F_OFF)      { W = W1; kcn = KC1; rem = idx - W1F_OFF; kdim = DIN;  }
  else if (idx < W3F_OFF) { W = W2; kcn = KC2; rem = idx - W2F_OFF; kdim = SIZE; }
  else if (idx < W4F_OFF) { W = W3; kcn = KC2; rem = idx - W3F_OFF; kdim = SIZE; }
  else                    { W = W4; kcn = KC2; rem = idx - W4F_OFF; kdim = SIZE; }
  int frag   = rem >> 9;
  int within = rem & 511;
  int lane   = within >> 4;
  int h      = within & 15;
  int nt     = frag / kcn;
  int kc     = frag % kcn;
  int k = kc * 32 + ((h >> 3) << 4) + ((lane >> 4) << 3) + (h & 7);
  int n = nt * 16 + (lane & 15);
  float v = (k < kdim) ? W[k * SIZE + n] : 0.0f;
  wfrag[idx] = (_Float16)v;
}

__global__ void rn_zero_out(float* __restrict__ out, int n) {
  int i = blockIdx.x * blockDim.x + threadIdx.x;
  if (i < n) out[i] = 0.0f;
}

// ---------------------------------------------------------------------------
// TDM: stage one layer's B-fragment panel (KC * 16KB, contiguous) into lds_w.
// 2D descriptor per ISA 08_async_tensor §8.3/8.4:
//   data_size = 8B (code 3); row = 2048 elems = 16KB; tile/tensor = 2048 x KC.
// Issued by one wave; completion via TENSORcnt.
// amdgpu-toolchain (clang-23) builtin arity: 6 args, extra int32x8 group.
// ---------------------------------------------------------------------------
__device__ __forceinline__ void tdm_stage_weights(const _Float16* gw,
                                                  unsigned lds_byte_off,
                                                  int kc_rows) {
  unsigned long long ga = (unsigned long long)(uintptr_t)gw;
  u32x4 g0;
  g0[0] = 1u;                                        // count=1, user descriptor
  g0[1] = lds_byte_off;                              // lds_addr
  g0[2] = (unsigned)(ga & 0xFFFFFFFFu);              // global_addr[31:0]
  g0[3] = (unsigned)((ga >> 32) & 0x01FFFFFFu)       // global_addr[56:32]
          | (2u << 30);                              // type=2 ("image")
  i32x8 g1;
  g1[0] = (int)(3u << 16);                           // data_size=3 (8B), mask=0
  g1[1] = (int)(2048u << 16);                        // [63:48] tensor_dim0 lo16
  g1[2] = (int)((unsigned)kc_rows << 16);            // [95:80] tensor_dim1
  g1[3] = (int)(2048u << 16);                        // [127:112] tile_dim0
  g1[4] = kc_rows;                                   // [143:128] tile_dim1
  g1[5] = 2048;                                      // tensor_dim0_stride lo32
  g1[6] = 0;
  g1[7] = 0;                                         // dim1 stride unused (2D)
  i32x4 gz4 = {0, 0, 0, 0};
  i32x8 gz8 = {0, 0, 0, 0, 0, 0, 0, 0};
  __builtin_amdgcn_tensor_load_to_lds(g0, g1, gz4, gz4, gz8, 0);
}

// ---------------------------------------------------------------------------
// One MLP layer: lds_h (f16 activations, HSTRIDE row stride) -> lds_h or out.
// KC = number of K32 chunks. FINAL: relu + per-batch sum via atomics.
// Two N-tiles per iteration -> 4 independent WMMA accumulation chains.
// ---------------------------------------------------------------------------
template <int KC, bool FINAL>
__device__ __forceinline__ void mlp_layer(const _Float16* __restrict__ gw,
                                          const float* __restrict__ bias,
                                          _Float16* lds_h, _Float16* lds_w,
                                          float* __restrict__ out, int bidx) {
  const int tid  = threadIdx.x;
  const int lane = tid & 31;
  const int wv   = tid >> 5;

  __syncthreads();   // prev layer fully done (lds_h written, lds_w free)

  // kick off async weight staging (wave 0), overlap with A gathers below
  if (wv == 0) {
    const unsigned lds_w_off =
        __builtin_amdgcn_groupstaticsize() + (unsigned)LDSH_BYTES;
    tdm_stage_weights(gw, lds_w_off, KC);
  }

  // gather A fragments (ISA 16-bit A 16x32 layout) for this wave's 2 tiles
  v16h A[TILES_PER_WAVE][KC];
  const int rbase = wv * (TILES_PER_WAVE * 16);
  const int koff  = (lane >> 4) << 3;           // 0 or 8
#pragma unroll
  for (int t = 0; t < TILES_PER_WAVE; ++t) {
    const int row = rbase + t * 16 + (lane & 15);
    const _Float16* hp = &lds_h[row * HSTRIDE];
#pragma unroll
    for (int kc = 0; kc < KC; ++kc) {
      const _Float16* p = hp + kc * 32 + koff;
      v8h lo = *(const v8h*)(p);                // K = kc*32 + koff + 0..7
      v8h hi = *(const v8h*)(p + 16);           // K = kc*32 + 16 + koff + 0..7
      v16h a;
#pragma unroll
      for (int i = 0; i < 8; ++i) { a[i] = lo[i]; a[8 + i] = hi[i]; }
      A[t][kc] = a;
    }
  }

  if (wv == 0) __builtin_amdgcn_s_wait_tensorcnt(0);  // DMA landed
  __syncthreads();   // weights visible; all lds_h reads done -> safe to rewrite

  const int ncol = lane & 15;
#pragma unroll 1
  for (int nt = 0; nt < NT; nt += 2) {
    v8f acc[2][TILES_PER_WAVE];
#pragma unroll
    for (int nn = 0; nn < 2; ++nn) {
      const float bv = bias[(nt + nn) * 16 + ncol];
#pragma unroll
      for (int t = 0; t < TILES_PER_WAVE; ++t)
#pragma unroll
        for (int r = 0; r < 8; ++r) acc[nn][t][r] = bv;
    }

#pragma unroll
    for (int kc = 0; kc < KC; ++kc) {
      v16h B0 = *(const v16h*)(&lds_w[((nt    ) * KC + kc) * 512 + lane * 16]);
      v16h B1 = *(const v16h*)(&lds_w[((nt + 1) * KC + kc) * 512 + lane * 16]);
#pragma unroll
      for (int t = 0; t < TILES_PER_WAVE; ++t)
        acc[0][t] = __builtin_amdgcn_wmma_f32_16x16x32_f16(
            false, A[t][kc], false, B0, (short)0, acc[0][t], false, false);
#pragma unroll
      for (int t = 0; t < TILES_PER_WAVE; ++t)
        acc[1][t] = __builtin_amdgcn_wmma_f32_16x16x32_f16(
            false, A[t][kc], false, B1, (short)0, acc[1][t], false, false);
    }

    // ReLU in one VALU op per element: med3(x, 0, +inf)
#pragma unroll
    for (int nn = 0; nn < 2; ++nn)
#pragma unroll
      for (int t = 0; t < TILES_PER_WAVE; ++t)
#pragma unroll
        for (int r = 0; r < 8; ++r)
          acc[nn][t][r] =
              __builtin_amdgcn_fmed3f(acc[nn][t][r], 0.0f, __builtin_inff());

    if (FINAL) {
      // D layout: lane<16 -> rows r (M=r), lane>=16 -> rows 8+r; col = ncol.
#pragma unroll
      for (int nn = 0; nn < 2; ++nn) {
        float s = 0.0f;
#pragma unroll
        for (int t = 0; t < TILES_PER_WAVE; ++t)
#pragma unroll
          for (int r = 0; r < 8; ++r) s += acc[nn][t][r];
        s += __shfl_xor(s, 16, 32);             // fold both half-tiles
        if (lane < 16)
          atomicAdd(&out[bidx * OUTC + (nt + nn) * 16 + ncol], s);
      }
    } else {
      const int rofs = (lane >> 4) << 3;        // 0 or 8
#pragma unroll
      for (int nn = 0; nn < 2; ++nn)
#pragma unroll
        for (int t = 0; t < TILES_PER_WAVE; ++t) {
          _Float16* hp =
              &lds_h[(rbase + t * 16 + rofs) * HSTRIDE + (nt + nn) * 16 + ncol];
#pragma unroll
          for (int r = 0; r < 8; ++r) hp[r * HSTRIDE] = (_Float16)acc[nn][t][r];
        }
    }
  }
}

// ---------------------------------------------------------------------------
// Fused 4-layer relation MLP. One WG = 256 pair-rows (one batch), 8 waves,
// each wave owns two 16-row tiles end-to-end.
// ---------------------------------------------------------------------------
__global__ __launch_bounds__(THREADS) void rn_mlp_kernel(
    const float* __restrict__ in_,
    const float* __restrict__ b1, const float* __restrict__ b2,
    const float* __restrict__ b3, const float* __restrict__ b4,
    const _Float16* __restrict__ wfrag,
    float* __restrict__ out) {
  extern __shared__ _Float16 smem[];
  _Float16* lds_h = smem;                          // 256 * 264 halfs
  _Float16* lds_w = smem + ROWS_PER_WG * HSTRIDE;  // 16*8*512 halfs

  const int tid = threadIdx.x;
  const long pbase = (long)blockIdx.x * ROWS_PER_WG;
  const int bidx = (int)(pbase / (NUM * NUM));

  // build layer-1 input row: [in[b,j], j, in[b,a], a, 0-pad] in f16
  {
    const long p = pbase + tid;
    const int j = (int)(p % NUM);
    const int a = (int)((p / NUM) % NUM);
    const int b = (int)(p / (NUM * NUM));
    const float* __restrict__ xj = in_ + (long)(b * NUM + j) * LENTH;
    const float* __restrict__ xa = in_ + (long)(b * NUM + a) * LENTH;
    _Float16* d = &lds_h[tid * HSTRIDE];
#pragma unroll 8
    for (int l = 0; l < LENTH; ++l) d[l] = (_Float16)xj[l];
    d[LENTH] = (_Float16)(float)j;
#pragma unroll 8
    for (int l = 0; l < LENTH; ++l) d[LENTH + 1 + l] = (_Float16)xa[l];
    d[2 * LENTH + 1] = (_Float16)(float)a;
#pragma unroll
    for (int l = DIN; l < K1PAD; ++l) d[l] = (_Float16)0.0f;
  }

  mlp_layer<KC1, false>(wfrag + W1F_OFF, b1, lds_h, lds_w, out, bidx);
  mlp_layer<KC2, false>(wfrag + W2F_OFF, b2, lds_h, lds_w, out, bidx);
  mlp_layer<KC2, false>(wfrag + W3F_OFF, b3, lds_h, lds_w, out, bidx);
  mlp_layer<KC2, true >(wfrag + W4F_OFF, b4, lds_h, lds_w, out, bidx);
}

extern "C" void kernel_launch(void* const* d_in, const int* in_sizes, int n_in,
                              void* d_out, int out_size, void* d_ws, size_t ws_size,
                              hipStream_t stream) {
  const float* in_ = (const float*)d_in[0];
  const float* W1  = (const float*)d_in[1];
  const float* b1  = (const float*)d_in[2];
  const float* W2  = (const float*)d_in[3];
  const float* b2  = (const float*)d_in[4];
  const float* W3  = (const float*)d_in[5];
  const float* b3  = (const float*)d_in[6];
  const float* W4  = (const float*)d_in[7];
  const float* b4  = (const float*)d_in[8];
  float* out = (float*)d_out;
  _Float16* wfrag = (_Float16*)d_ws;   // needs ~475 KB of scratch

  rn_pack_weights<<<(WF_TOTAL + 255) / 256, 256, 0, stream>>>(W1, W2, W3, W4, wfrag);
  rn_zero_out<<<(MB * OUTC + 255) / 256, 256, 0, stream>>>(out, MB * OUTC);

  const size_t smem_bytes =
      (size_t)(ROWS_PER_WG * HSTRIDE + NT * KC2 * 512) * sizeof(_Float16); // 260 KB
  (void)hipFuncSetAttribute((const void*)rn_mlp_kernel,
                            hipFuncAttributeMaxDynamicSharedMemorySize,
                            (int)smem_bytes);
  const int nblocks = (MB * NUM * NUM) / ROWS_PER_WG;  // 1024
  rn_mlp_kernel<<<nblocks, THREADS, smem_bytes, stream>>>(
      in_, b1, b2, b3, b4, wfrag, out);
}
